// PiTWithCoords_71382356460017
// MI455X (gfx1250) — matrix operand order repaired and checked by reading, not compiled, not run on hardware
//
#include <hip/hip_runtime.h>
#include <hip/hip_bf16.h>
#include <math.h>

// ---------------------------------------------------------------------------
// PiT-with-coords forward for MI455X (gfx1250, wave32, WMMA).
// All heavy matmuls use v_wmma_f32_16x16x32_f16 with f16 operands stored in
// pre-packed per-lane fragment layouts (32 lanes x 16 f16 per 16x32 A-tile /
// 32x16 B-tile), so GEMM inner loops are pure b128 loads + WMMA.
// att_gemm (dominant kernel) is software-pipelined: next iteration's
// fragments are loaded before the current WMMAs consume their registers.
// ---------------------------------------------------------------------------

typedef __attribute__((ext_vector_type(16))) _Float16 v16h;
typedef __attribute__((ext_vector_type(8)))  float    v8f;

#define HIDN  128
#define NHEAD 8
#define VDIM  16
#define BB    16          // batch
#define NIN   4096        // 64*64
#define NLAT  1024        // 32*32
#define M_IN  (BB*NIN)    // 65536
#define M_LAT (BB*NLAT)   // 16384

__device__ __forceinline__ float gelu_f(float x) {
  float x3 = x * x * x;
  return 0.5f * x * (1.0f + tanhf(0.7978845608f * (x + 0.044715f * x3)));
}

// Packed-A fragment layout for V_WMMA_F32_16X16X32_F16 (16x32 f16 tile):
// lane = 16*half + (m%16); inverse mapping (m,k) -> flat element offset.
__device__ __forceinline__ size_t pa_off(int m, int k, int K) {
  int mt = m >> 4, kt = k >> 5, r = m & 15, kk = k & 31;
  int half = (kk >> 3) & 1;
  int e = (kk & 7) | (((kk >> 4) & 1) << 3);
  int lane = (half << 4) | r;
  return ((size_t)(mt * (K >> 5) + kt) * 32 + lane) * 16 + e;
}

// Packed-B fragment layout (32x16 f16 tile): lane = 16*((kk>>3)&1) + (n%16),
// element e = (kk & 7) | (((kk >> 4) & 1) << 3).
__device__ __forceinline__ size_t pb_off(int k, int n, int N) {
  int kt = k >> 5, nt = n >> 4, kk = k & 31;
  int half = (kk >> 3) & 1, e = (kk & 7) | (((kk >> 4) & 1) << 3);
  int lane = (half << 4) | (n & 15);
  return ((size_t)(kt * (N >> 4) + nt) * 32 + lane) * 16 + e;
}

// ------------------------- weight packing ----------------------------------
__global__ void pack_plain(const float* __restrict__ W, _Float16* __restrict__ PB) {
  int t = blockIdx.x * blockDim.x + threadIdx.x;   // 16384 threads
  int k = t >> 7, n = t & 127;
  PB[pb_off(k, n, HIDN)] = (_Float16)W[t];
}

// W f32 [8,128,16] (head,hid,v) -> concat-head packed-B f16 [128,128]
__global__ void pack_headcat(const float* __restrict__ W, _Float16* __restrict__ PB) {
  int t = blockIdx.x * blockDim.x + threadIdx.x;
  int k = t >> 7, n = t & 127;
  int h = n >> 4, v = n & 15;
  PB[pb_off(k, n, HIDN)] = (_Float16)W[((size_t)h * HIDN + k) * VDIM + v];
}

// ------------------------- encoder (K=3, VALU) -----------------------------
__global__ void encoder_kernel(const float* __restrict__ x, const float* __restrict__ W,
                               const float* __restrict__ bias, _Float16* __restrict__ H0) {
  int t = blockIdx.x * blockDim.x + threadIdx.x;   // M_IN*128 threads
  int m = t >> 7, c = t & 127;
  float a = bias[c] + x[3 * m] * W[c] + x[3 * m + 1] * W[HIDN + c] + x[3 * m + 2] * W[2 * HIDN + c];
  H0[pa_off(m, c, HIDN)] = (_Float16)gelu_f(a);
}

// -------------------- 2%-percentile threshold (proc) -----------------------
__global__ void thr_kernel(float* __restrict__ thr) {
  int q = blockIdx.x * blockDim.x + threadIdx.x;
  if (q >= NLAT) return;
  const float inv = 1.0f / 32.0f;
  float qx = (float)(q & 31) * inv, qy = (float)(q >> 5) * inv;
  float best[22];
#pragma unroll
  for (int i = 0; i < 22; i++) best[i] = 3.0e38f;
  for (int k = 0; k < NLAT; k++) {
    float kx = (float)(k & 31) * inv, ky = (float)(k >> 5) * inv;
    float dx = qx - kx, dy = qy - ky;
    float d = 0.5f * (dx * dx + dy * dy);
    if (d < best[21]) {
      int j = 21;
      while (j > 0 && best[j - 1] > d) { best[j] = best[j - 1]; --j; }
      best[j] = d;
    }
  }
  // pos = 0.02*(1024-1) = 20.46 -> v[20] + 0.46*(v[21]-v[20])
  const float frac = 0.02f * 1023.0f - 20.0f;
  thr[q] = best[20] + frac * (best[21] - best[20]);
}

// -------------------- attention generation (softmax) -----------------------
__global__ void att_gen(const float* __restrict__ r8, const float* __restrict__ thr,
                        _Float16* __restrict__ attP, int Nq, int Nk,
                        int qrx, int krx, float invq, float invk) {
  __shared__ float red[256];
  int q = blockIdx.x, h = blockIdx.y, tid = threadIdx.x;
  float scale = tanf(0.78539808f * (1.0f + sinf(r8[h])));  // 0.25*pi*(1-1e-7)
  float qx = (float)(q % qrx) * invq, qy = (float)(q / qrx) * invq;
  float thrq = thr ? thr[q] : 3.4e38f;

  float mn = 3.4e38f;
  for (int k = tid; k < Nk; k += 256) {
    float kx = (float)(k % krx) * invk, ky = (float)(k / krx) * invk;
    float dx = qx - kx, dy = qy - ky;
    float d = 0.5f * (dx * dx + dy * dy);
    if (d <= thrq) mn = fminf(mn, d);
  }
  red[tid] = mn; __syncthreads();
  for (int s = 128; s > 0; s >>= 1) { if (tid < s) red[tid] = fminf(red[tid], red[tid + s]); __syncthreads(); }
  float dmin = red[0]; __syncthreads();

  float sm = 0.0f;
  for (int k = tid; k < Nk; k += 256) {
    float kx = (float)(k % krx) * invk, ky = (float)(k / krx) * invk;
    float dx = qx - kx, dy = qy - ky;
    float d = 0.5f * (dx * dx + dy * dy);
    if (d <= thrq) sm += expf(scale * (dmin - d));
  }
  red[tid] = sm; __syncthreads();
  for (int s = 128; s > 0; s >>= 1) { if (tid < s) red[tid] += red[tid + s]; __syncthreads(); }
  float invs = 1.0f / red[0];

  _Float16* base = attP + (size_t)h * (size_t)Nq * (size_t)Nk;
  for (int k = tid; k < Nk; k += 256) {
    float kx = (float)(k % krx) * invk, ky = (float)(k / krx) * invk;
    float dx = qx - kx, dy = qy - ky;
    float d = 0.5f * (dx * dx + dy * dy);
    float a = (d <= thrq) ? expf(scale * (dmin - d)) * invs : 0.0f;
    base[pa_off(q, k, Nk)] = (_Float16)a;
  }
}

// -------------------- value projection GEMM --------------------------------
__global__ void __launch_bounds__(256) value_gemm(const _Float16* __restrict__ Ap,
                                                  const _Float16* __restrict__ PW,
                                                  _Float16* __restrict__ valr,
                                                  int M, int Nk) {
  int lane = threadIdx.x & 31;
  int w = (blockIdx.x * blockDim.x + threadIdx.x) >> 5;
  int total = (M >> 4) * 2;
  if (w >= total) return;
  int mt = w >> 1, nt0 = (w & 1) * 4;
  v8f acc[4] = {};
#pragma unroll
  for (int kt = 0; kt < 4; kt++) {
    v16h a = *(const v16h*)(Ap + ((size_t)(mt * 4 + kt) * 32 + lane) * 16);
#pragma unroll
    for (int j = 0; j < 4; j++) {
      v16h b = *(const v16h*)(PW + ((size_t)(kt * 8 + nt0 + j) * 32 + lane) * 16);
      acc[j] = __builtin_amdgcn_wmma_f32_16x16x32_f16(false, a, false, b, (short)0, acc[j], false, false);
    }
  }
  int v = lane & 15, hl = lane >> 4;
#pragma unroll
  for (int j = 0; j < 4; j++) {
    int hh = nt0 + j;
    _Float16* dst = valr + (size_t)hh * (size_t)Nk * 256;
#pragma unroll
    for (int i = 0; i < 8; i++) {
      int m = (mt << 4) + i + (hl << 3);
      int n = m % Nk, b = m / Nk;
      dst[pb_off(n, b * VDIM + v, 256)] = (_Float16)acc[j][i];
    }
  }
}

// -------------------- attention x value GEMM (+gelu) -----------------------
// Per head: att[h] [Nq,Nk] (packed-A) x valr[h] [Nk,256] (packed-B) -> gelu
// -> Hout packed-A [B*Nq, 128].  Software-pipelined K-loop.
__global__ void __launch_bounds__(256) att_gemm(const _Float16* __restrict__ attP,
                                                const _Float16* __restrict__ valr,
                                                _Float16* __restrict__ Hout,
                                                int Nq, int Nk) {
  int lane = threadIdx.x & 31;
  int w = (blockIdx.x * blockDim.x + threadIdx.x) >> 5;
  int gph = (Nq >> 4) * 4;           // groups per head (4 col-tiles each)
  if (w >= NHEAD * gph) return;
  int hh = w / gph, rem = w % gph;
  int mt = rem >> 2, nt0 = (rem & 3) * 4;
  int KT = Nk >> 5;
  // per-lane base pointers; A tile stride = 512 elem, B kt-stride = 16*512
  const _Float16* ap = attP + (size_t)hh * (size_t)Nq * (size_t)Nk
                     + ((size_t)mt * KT * 32 + lane) * 16;
  const _Float16* bp = valr + (size_t)hh * (size_t)Nk * 256
                     + ((size_t)nt0 * 32 + lane) * 16;

  v8f acc0 = {}, acc1 = {}, acc2 = {}, acc3 = {};
  v16h a  = *(const v16h*)ap;
  v16h b0 = *(const v16h*)(bp);
  v16h b1 = *(const v16h*)(bp + 512);
  v16h b2 = *(const v16h*)(bp + 1024);
  v16h b3 = *(const v16h*)(bp + 1536);
  for (int kt = 0; kt < KT - 1; ++kt) {
    // prefetch next iteration's fragments before consuming current ones
    const _Float16* apn = ap + (size_t)(kt + 1) * 512;
    const _Float16* bpn = bp + (size_t)(kt + 1) * 8192;
    v16h an  = *(const v16h*)apn;
    v16h bn0 = *(const v16h*)(bpn);
    v16h bn1 = *(const v16h*)(bpn + 512);
    v16h bn2 = *(const v16h*)(bpn + 1024);
    v16h bn3 = *(const v16h*)(bpn + 1536);
    acc0 = __builtin_amdgcn_wmma_f32_16x16x32_f16(false, a, false, b0, (short)0, acc0, false, false);
    acc1 = __builtin_amdgcn_wmma_f32_16x16x32_f16(false, a, false, b1, (short)0, acc1, false, false);
    acc2 = __builtin_amdgcn_wmma_f32_16x16x32_f16(false, a, false, b2, (short)0, acc2, false, false);
    acc3 = __builtin_amdgcn_wmma_f32_16x16x32_f16(false, a, false, b3, (short)0, acc3, false, false);
    a = an; b0 = bn0; b1 = bn1; b2 = bn2; b3 = bn3;
  }
  acc0 = __builtin_amdgcn_wmma_f32_16x16x32_f16(false, a, false, b0, (short)0, acc0, false, false);
  acc1 = __builtin_amdgcn_wmma_f32_16x16x32_f16(false, a, false, b1, (short)0, acc1, false, false);
  acc2 = __builtin_amdgcn_wmma_f32_16x16x32_f16(false, a, false, b2, (short)0, acc2, false, false);
  acc3 = __builtin_amdgcn_wmma_f32_16x16x32_f16(false, a, false, b3, (short)0, acc3, false, false);

  int v = lane & 15, hl = lane >> 4;
  v8f accs[4] = {acc0, acc1, acc2, acc3};
#pragma unroll
  for (int j = 0; j < 4; j++) {
    int b = nt0 + j;                 // column-tile index == batch
#pragma unroll
    for (int i = 0; i < 8; i++) {
      int q = (mt << 4) + i + (hl << 3);
      int row = b * Nq + q, col = hh * VDIM + v;
      Hout[pa_off(row, col, HIDN)] = (_Float16)gelu_f(accs[j][i]);
    }
  }
}

// -------------------- generic hidden GEMM [M,128]x[128,128] ----------------
__global__ void __launch_bounds__(256) gemm_hidden(const _Float16* __restrict__ Ap,
                                                   const _Float16* __restrict__ PB,
                                                   const float* __restrict__ bias,
                                                   const float* __restrict__ addend,
                                                   float* __restrict__ outF,
                                                   _Float16* __restrict__ outP,
                                                   int M, int do_gelu) {
  int lane = threadIdx.x & 31;
  int w = (blockIdx.x * blockDim.x + threadIdx.x) >> 5;
  int total = (M >> 4) * 2;
  if (w >= total) return;
  int mt = w >> 1, nt0 = (w & 1) * 4;
  v8f acc[4] = {};
#pragma unroll
  for (int kt = 0; kt < 4; kt++) {
    v16h a = *(const v16h*)(Ap + ((size_t)(mt * 4 + kt) * 32 + lane) * 16);
#pragma unroll
    for (int j = 0; j < 4; j++) {
      v16h b = *(const v16h*)(PB + ((size_t)(kt * 8 + nt0 + j) * 32 + lane) * 16);
      acc[j] = __builtin_amdgcn_wmma_f32_16x16x32_f16(false, a, false, b, (short)0, acc[j], false, false);
    }
  }
  int cl = lane & 15, hl = lane >> 4;
#pragma unroll
  for (int j = 0; j < 4; j++) {
    int c = (nt0 + j) * 16 + cl;
    float bv = bias ? bias[c] : 0.0f;
#pragma unroll
    for (int i = 0; i < 8; i++) {
      int mrow = (mt << 4) + i + (hl << 3);
      float val = acc[j][i] + bv;
      if (addend) val += addend[(size_t)mrow * HIDN + c];
      if (do_gelu) val = gelu_f(val);
      if (outP) outP[pa_off(mrow, c, HIDN)] = (_Float16)val;
      else      outF[(size_t)mrow * HIDN + c] = val;
    }
  }
}

// -------------------- final 128 -> 1 projection ----------------------------
__global__ void final_kernel(const _Float16* __restrict__ Hp, const float* __restrict__ w2,
                             const float* __restrict__ b2, float* __restrict__ out, int M) {
  int t = blockIdx.x * blockDim.x + threadIdx.x;
  if (t >= M) return;
  float acc = b2[0];
#pragma unroll 4
  for (int j = 0; j < HIDN; j++) acc += (float)Hp[pa_off(t, j, HIDN)] * w2[j];
  out[t] = acc;
}

// ===========================================================================
extern "C" void kernel_launch(void* const* d_in, const int* in_sizes, int n_in,
                              void* d_out, int out_size, void* d_ws, size_t ws_size,
                              hipStream_t stream) {
  (void)in_sizes; (void)n_in; (void)out_size; (void)ws_size;
  const float* x      = (const float*)d_in[0];
  const float* en_w   = (const float*)d_in[1];
  const float* en_b   = (const float*)d_in[2];
  const float* down_r = (const float*)d_in[3];
  const float* down_w = (const float*)d_in[4];
  const float* pa_r   = (const float*)d_in[5];
  const float* pa_w   = (const float*)d_in[6];
  const float* mlp1_w = (const float*)d_in[7];
  const float* mlp1_b = (const float*)d_in[8];
  const float* mlp2_w = (const float*)d_in[9];
  const float* mlp2_b = (const float*)d_in[10];
  const float* res_w  = (const float*)d_in[11];
  const float* res_b  = (const float*)d_in[12];
  const float* up_r   = (const float*)d_in[13];
  const float* up_w   = (const float*)d_in[14];
  const float* de1_w  = (const float*)d_in[15];
  const float* de1_b  = (const float*)d_in[16];
  const float* de2_w  = (const float*)d_in[17];
  const float* de2_b  = (const float*)d_in[18];
  float* out = (float*)d_out;

  // ---- workspace carve-up (~143 MB) ----
  char* ws = (char*)d_ws;
  size_t off = 0;
  auto take = [&](size_t sz) -> char* { char* p = ws + off; off = (off + sz + 255) & ~(size_t)255; return p; };
  const size_t SZ_PW = (size_t)HIDN * HIDN * 2;
  _Float16* PW_down = (_Float16*)take(SZ_PW);
  _Float16* PW_pa[3], *PW_m1[3], *PW_m2[3], *PW_rs[3];
  for (int i = 0; i < 3; i++) PW_pa[i] = (_Float16*)take(SZ_PW);
  for (int i = 0; i < 3; i++) PW_m1[i] = (_Float16*)take(SZ_PW);
  for (int i = 0; i < 3; i++) PW_m2[i] = (_Float16*)take(SZ_PW);
  for (int i = 0; i < 3; i++) PW_rs[i] = (_Float16*)take(SZ_PW);
  _Float16* PW_up  = (_Float16*)take(SZ_PW);
  _Float16* PW_de1 = (_Float16*)take(SZ_PW);
  float*    thrB   = (float*)take(NLAT * 4);
  _Float16* h0     = (_Float16*)take((size_t)M_IN * HIDN * 2);   // reused post-decoder-fc1
  _Float16* hU     = (_Float16*)take((size_t)M_IN * HIDN * 2);
  _Float16* hL1    = (_Float16*)take((size_t)M_LAT * HIDN * 2);
  _Float16* hL2    = (_Float16*)take((size_t)M_LAT * HIDN * 2);
  _Float16* paB    = (_Float16*)take((size_t)M_LAT * HIDN * 2);
  _Float16* t1B    = (_Float16*)take((size_t)M_LAT * HIDN * 2);
  float*    resF   = (float*)take((size_t)M_LAT * HIDN * 4);
  _Float16* valr   = (_Float16*)take((size_t)NHEAD * NIN * 256 * 2);
  _Float16* attB   = (_Float16*)take((size_t)NHEAD * NLAT * NIN * 2); // reused down/proc/up

  // ---- weight packing (f32 -> packed-B f16) ----
  pack_headcat<<<64, 256, 0, stream>>>(down_w, PW_down);
  for (int i = 0; i < 3; i++) {
    pack_headcat<<<64, 256, 0, stream>>>(pa_w + (size_t)i * NHEAD * HIDN * VDIM, PW_pa[i]);
    pack_plain  <<<64, 256, 0, stream>>>(mlp1_w + (size_t)i * HIDN * HIDN, PW_m1[i]);
    pack_plain  <<<64, 256, 0, stream>>>(mlp2_w + (size_t)i * HIDN * HIDN, PW_m2[i]);
    pack_plain  <<<64, 256, 0, stream>>>(res_w  + (size_t)i * HIDN * HIDN, PW_rs[i]);
  }
  pack_headcat<<<64, 256, 0, stream>>>(up_w, PW_up);
  pack_plain  <<<64, 256, 0, stream>>>(de1_w, PW_de1);

  // ---- encoder + locality threshold ----
  encoder_kernel<<<(M_IN * HIDN) / 256, 256, 0, stream>>>(x, en_w, en_b, h0);
  thr_kernel<<<NLAT / 256, 256, 0, stream>>>(thrB);

  // ---- pos_att DOWN: [B,4096,128] -> [B,1024,128] (loc=103: dense) ----
  value_gemm<<<((M_IN / 16) * 2) / 8, 256, 0, stream>>>(h0, PW_down, valr, M_IN, NIN);
  att_gen<<<dim3(NLAT, NHEAD), 256, 0, stream>>>(down_r, nullptr, attB, NLAT, NIN,
                                                 32, 64, 1.0f / 32.0f, 1.0f / 64.0f);
  att_gemm<<<(NHEAD * (NLAT / 16) * 4) / 8, 256, 0, stream>>>(attB, valr, hL1, NLAT, NIN);

  // ---- processor blocks (loc=2: sparse mask, head-independent thr) ----
  _Float16* hcur = hL1; _Float16* hnext = hL2;
  const int gLat = ((M_LAT / 16) * 2) / 8;   // 256 blocks
  for (int i = 0; i < 3; i++) {
    value_gemm<<<gLat, 256, 0, stream>>>(hcur, PW_pa[i], valr, M_LAT, NLAT);
    att_gen<<<dim3(NLAT, NHEAD), 256, 0, stream>>>(pa_r + i * NHEAD, thrB, attB, NLAT, NLAT,
                                                   32, 32, 1.0f / 32.0f, 1.0f / 32.0f);
    att_gemm<<<(NHEAD * (NLAT / 16) * 4) / 8, 256, 0, stream>>>(attB, valr, paB, NLAT, NLAT);
    gemm_hidden<<<gLat, 256, 0, stream>>>(paB, PW_m1[i], mlp1_b + i * HIDN, nullptr,
                                          nullptr, t1B, M_LAT, 1);
    gemm_hidden<<<gLat, 256, 0, stream>>>(hcur, PW_rs[i], res_b + i * HIDN, nullptr,
                                          resF, nullptr, M_LAT, 0);
    gemm_hidden<<<gLat, 256, 0, stream>>>(t1B, PW_m2[i], mlp2_b + i * HIDN, resF,
                                          nullptr, hnext, M_LAT, 1);
    _Float16* tmp = hcur; hcur = hnext; hnext = tmp;
  }

  // ---- pos_att UP: [B,1024,128] -> [B,4096,128] (dense) ----
  value_gemm<<<gLat, 256, 0, stream>>>(hcur, PW_up, valr, M_LAT, NLAT);
  att_gen<<<dim3(NIN, NHEAD), 256, 0, stream>>>(up_r, nullptr, attB, NIN, NLAT,
                                                64, 32, 1.0f / 64.0f, 1.0f / 32.0f);
  att_gemm<<<(NHEAD * (NIN / 16) * 4) / 8, 256, 0, stream>>>(attB, valr, hU, NIN, NLAT);

  // ---- decoder ----
  gemm_hidden<<<((M_IN / 16) * 2) / 8, 256, 0, stream>>>(hU, PW_de1, de1_b, nullptr,
                                                         nullptr, h0, M_IN, 1);
  final_kernel<<<M_IN / 256, 256, 0, stream>>>(h0, de2_w, de2_b, out, M_IN);
}
